// RobustSplatAttention_43224550867589
// MI455X (gfx1250) — compile-verified, matrix operand order checked by reading, not run on hardware
//
#include <hip/hip_runtime.h>

// ---------------------------------------------------------------------------
// RobustSplatAttention on MI455X (gfx1250, wave32, WMMA + double-buffered TDM)
// B=2, S=2048, D=512, H=8, K=16, DH=64
// attn is rank-16 (attn = wq @ aff_k^T) -> S x S matrix never formed.
// Heavy work = four 4096x512x512 GEMMs: TDM tile prefetch (2-deep pipeline)
// -> LDS -> ds_load_b128 fragments -> v_wmma_f32_16x16x32_bf16.
// ---------------------------------------------------------------------------

#define Bq   2
#define Sq   2048
#define Dq   512
#define Hq   8
#define Kq   16
#define DHq  64
#define MROWS (Bq * Sq)          // 4096
#define KSTEPS (Dq / 32)         // 16
#define EPSF 1e-8f

typedef __attribute__((ext_vector_type(16))) __bf16        bf16x16;
typedef __attribute__((ext_vector_type(8)))  float         f32x8;
typedef __attribute__((ext_vector_type(4)))  unsigned int  u32x4;
typedef __attribute__((ext_vector_type(8)))  int           i32x8;
typedef __attribute__((ext_vector_type(4)))  int           i32x4;

#if defined(__has_builtin)
#if __has_builtin(__builtin_amdgcn_tensor_load_to_lds) && \
    __has_builtin(__builtin_amdgcn_s_wait_tensorcnt)
#define USE_TDM 1
#endif
#endif
#ifndef USE_TDM
#define USE_TDM 0
#endif

// ---------------------------------------------------------------------------
// Stage 1a: f32 -> bf16 elementwise
// ---------------------------------------------------------------------------
__global__ void convert_x_kernel(const float* __restrict__ src,
                                 __bf16* __restrict__ dst, int n) {
  int i = blockIdx.x * blockDim.x + threadIdx.x;
  if (i < n) dst[i] = (__bf16)src[i];
}

// ---------------------------------------------------------------------------
// Stage 1b: transpose + convert the 4 weight matrices (so B-fragments of the
// WMMA GEMM have contiguous k: Bt[n][k] = W[k][n]).  grid.z selects matrix.
// ---------------------------------------------------------------------------
__global__ void convert_w_kernel(const float* __restrict__ W0,
                                 const float* __restrict__ W1,
                                 const float* __restrict__ W2,
                                 const float* __restrict__ W3,
                                 __bf16* __restrict__ Wt) {
  int z = blockIdx.z;
  const float* W = (z == 0) ? W0 : (z == 1) ? W1 : (z == 2) ? W2 : W3;
  int idx = blockIdx.x * blockDim.x + threadIdx.x;   // over 512*512
  int r = idx >> 9;          // row in W  (k dim)
  int c = idx & 511;         // col in W  (n dim)
  Wt[(size_t)z * Dq * Dq + (size_t)c * Dq + r] = (__bf16)W[idx];
}

// ---------------------------------------------------------------------------
// TDM: issue one D# for a 32-row x 32-k bf16 tile -> LDS (row-major, packed).
// Descriptor per CDNA5 ISA sec 8.3/8.4.  All fields wave-uniform -> SGPRs.
// ---------------------------------------------------------------------------
#if USE_TDM
__device__ __forceinline__ void tdm_load_tile32x32(const __bf16* gsrc,
                                                   unsigned lds_bytes,
                                                   int tensor_rows) {
  unsigned long long ga = (unsigned long long)(uintptr_t)gsrc;
  u32x4 g0;
  g0[0] = 1u;                                          // count=1 (valid, user)
  g0[1] = lds_bytes;                                   // lds_addr
  g0[2] = (unsigned)ga;                                // global_addr[31:0]
  g0[3] = ((unsigned)(ga >> 32) & 0x01FFFFFFu)         // global_addr[56:32]
          | 0x80000000u;                               // type=2 (bits 127:126)
  i32x8 g1;
  g1[0] = (1 << 16);                                   // data_size=1 -> 2 bytes
  g1[1] = (Dq & 0xFFFF) << 16;                         // tensor_dim0 lo16 = 512
  g1[2] = (tensor_rows & 0xFFFF) << 16;                // dim0 hi16=0 | dim1 lo16
  g1[3] = (32 << 16) | ((tensor_rows >> 16) & 0xFFFF); // dim1 hi16 | tile_dim0=32
  g1[4] = 32;                                          // tile_dim1=32, tile_dim2=0
  g1[5] = Dq;                                          // tensor_dim0_stride lo32
  g1[6] = 0;
  g1[7] = 0;
  i32x4 z4 = {0, 0, 0, 0};
#if defined(__clang_major__) && (__clang_major__ >= 23)
  i32x8 z8 = {};
  __builtin_amdgcn_tensor_load_to_lds(g0, g1, z4, z4, z8, 0);
#else
  __builtin_amdgcn_tensor_load_to_lds(g0, g1, z4, z4, 0);
#endif
}
#endif

// ---------------------------------------------------------------------------
// WMMA GEMM: C[4096,512] = A[4096,512] * Bt[512,512]^T  (bf16 -> f32 accum)
// Block = 128 thr (4 waves), block tile 64x64, wave tile 32x32 (2x2 WMMA),
// K-step 32, TDM double-buffered 2-deep: tile i+1 streams in while tile i
// feeds the WMMA pipes.  s_wait_tensorcnt(1) retires the oldest DMA (TDM ops
// complete in order per wave).  Fragment layout per ISA (16-bit A 16x32):
// lane m=lane&15 holds row m, kg=lane>>4 selects K runs {8kg.., 16+8kg..}.
// ---------------------------------------------------------------------------
__device__ __forceinline__ bf16x16 frag_from_rows(const __bf16* rowbase,
                                                  int lane) {
  int m  = lane & 15;
  int kg = lane >> 4;
  const __bf16* p = rowbase + (size_t)m * 32 + 8 * kg;
  union { bf16x16 v; u32x4 u[2]; } t;
  t.u[0] = *(const u32x4*)(p);
  t.u[1] = *(const u32x4*)(p + 16);
  return t.v;
}

__device__ __forceinline__ void store_tile(float* __restrict__ C, int m0,
                                           int n0, int lane, f32x8 c) {
  int n = lane & 15;
  int half = lane >> 4;                       // lanes 16-31 hold M = r + 8
  float* p = C + (size_t)(m0 + 8 * half) * Dq + n0 + n;
#pragma unroll
  for (int r = 0; r < 8; ++r) p[(size_t)r * Dq] = c[r];
}

__global__ void __launch_bounds__(128)
wmma_gemm_kernel(const __bf16* __restrict__ A, const __bf16* __restrict__ Bt,
                 float* __restrict__ C, long strideB, long strideC) {
  __shared__ __attribute__((aligned(16))) __bf16 sA[2][64 * 32];  // 8 KB
  __shared__ __attribute__((aligned(16))) __bf16 sB[2][64 * 32];  // 8 KB

  const int lane = threadIdx.x & 31;
  const int wave = __builtin_amdgcn_readfirstlane((int)(threadIdx.x >> 5));
  const __bf16* Bz = Bt + (size_t)blockIdx.z * strideB;
  float*        Cz = C + (size_t)blockIdx.z * strideC;

  const int mblk = blockIdx.y * 64;
  const int nblk = blockIdx.x * 64;
  const int mw = (wave >> 1) * 32;            // wave tile within block
  const int nw = (wave & 1) * 32;

  f32x8 c00 = {}, c01 = {}, c10 = {}, c11 = {};

#if USE_TDM
  const unsigned ldsA = (unsigned)(uintptr_t)&sA[0][0];
  const unsigned ldsB = (unsigned)(uintptr_t)&sB[0][0];
  // each wave DMAs a disjoint 32x32 quarter: waves 0,1 -> A; 2,3 -> B
  const __bf16* gq = (wave < 2)
      ? A  + (size_t)(mblk + wave * 32) * Dq
      : Bz + (size_t)(nblk + (wave - 2) * 32) * Dq;
  const unsigned ldq = (wave < 2) ? (ldsA + (unsigned)wave * 2048u)
                                  : (ldsB + (unsigned)(wave - 2) * 2048u);
  const int trows = (wave < 2) ? MROWS : Dq;

  tdm_load_tile32x32(gq, ldq, trows);                 // prefetch tile 0
#endif

  for (int i = 0; i < KSTEPS; ++i) {
    const int buf = i & 1;
#if USE_TDM
    __syncthreads();             // everyone done reading buf^1 (iter i-1)
    if (i + 1 < KSTEPS) {
      tdm_load_tile32x32(gq + (i + 1) * 32, ldq + (unsigned)(buf ^ 1) * 4096u,
                         trows);
      __builtin_amdgcn_s_wait_tensorcnt(1);  // oldest (tile i) complete
    } else {
      __builtin_amdgcn_s_wait_tensorcnt(0);
    }
    __syncthreads();             // tile i published WGP-wide
#else
    __syncthreads();
    // fallback: cooperative vector copy (1024 dwords per matrix, 128 threads)
#pragma unroll
    for (int j = 0; j < 8; ++j) {
      int t = threadIdx.x + j * 128;          // dword index
      int row = t >> 4, cd = t & 15;
      ((unsigned*)sA[buf])[t] =
          *(const unsigned*)(A + (size_t)(mblk + row) * Dq + i * 32 + cd * 2);
      ((unsigned*)sB[buf])[t] =
          *(const unsigned*)(Bz + (size_t)(nblk + row) * Dq + i * 32 + cd * 2);
    }
    __syncthreads();
#endif

    bf16x16 a0 = frag_from_rows(&sA[buf][(mw + 0)  * 32], lane);
    bf16x16 a1 = frag_from_rows(&sA[buf][(mw + 16) * 32], lane);
    bf16x16 b0 = frag_from_rows(&sB[buf][(nw + 0)  * 32], lane);
    bf16x16 b1 = frag_from_rows(&sB[buf][(nw + 16) * 32], lane);
    c00 = __builtin_amdgcn_wmma_f32_16x16x32_bf16(false, a0, false, b0,
                                                  (short)0, c00, false, false);
    c01 = __builtin_amdgcn_wmma_f32_16x16x32_bf16(false, a0, false, b1,
                                                  (short)0, c01, false, false);
    c10 = __builtin_amdgcn_wmma_f32_16x16x32_bf16(false, a1, false, b0,
                                                  (short)0, c10, false, false);
    c11 = __builtin_amdgcn_wmma_f32_16x16x32_bf16(false, a1, false, b1,
                                                  (short)0, c11, false, false);
  }

  store_tile(Cz, mblk + mw,      nblk + nw,      lane, c00);
  store_tile(Cz, mblk + mw,      nblk + nw + 16, lane, c01);
  store_tile(Cz, mblk + mw + 16, nblk + nw,      lane, c10);
  store_tile(Cz, mblk + mw + 16, nblk + nw + 16, lane, c11);
}

// ---------------------------------------------------------------------------
// Stage 3: Gaussian splat affinity.
// T is q or k in [B*S, D] layout; output aff [B,H,S,K] (optionally * amp).
// grid = (S/256, H, B), 256 threads, one s per thread.
// ---------------------------------------------------------------------------
__global__ void __launch_bounds__(256)
splat_affinity_kernel(const float* __restrict__ T,
                      const float* __restrict__ pos,
                      const float* __restrict__ lsc,
                      const float* __restrict__ amp,
                      float* __restrict__ out, int use_amp) {
  __shared__ float s_pos[Kq * DHq];
  __shared__ float s_inv[Kq], s_p2[Kq], s_amp[Kq];
  const int h = blockIdx.y, b = blockIdx.z;
  const int tid = threadIdx.x;

  for (int i = tid; i < Kq * DHq; i += 256) s_pos[i] = pos[h * Kq * DHq + i];
  __syncthreads();
  if (tid < Kq) {
    float sc = __expf(lsc[h * Kq + tid]);
    sc = fminf(fmaxf(sc, 0.1f), 2.0f);
    s_inv[tid] = 1.0f / (2.0f * sc * sc);
    s_amp[tid] = use_amp ? amp[h * Kq + tid] : 1.0f;
    float p2 = 0.f;
#pragma unroll
    for (int d = 0; d < DHq; ++d) { float p = s_pos[tid * DHq + d]; p2 += p * p; }
    s_p2[tid] = p2;
  }
  __syncthreads();

  const int s = blockIdx.x * 256 + tid;
  const float* t = T + ((size_t)(b * Sq + s)) * Dq + h * DHq;
  float tr[DHq];
  float t2 = 0.f;
#pragma unroll
  for (int d = 0; d < DHq; ++d) { tr[d] = t[d]; t2 += tr[d] * tr[d]; }

  float* o = out + (((size_t)(b * Hq + h)) * Sq + s) * Kq;
  for (int k = 0; k < Kq; ++k) {
    float cr = 0.f;
#pragma unroll
    for (int d = 0; d < DHq; ++d) cr += tr[d] * s_pos[k * DHq + d];
    float d2 = fmaxf(t2 + s_p2[k] - 2.f * cr, 0.f);
    o[k] = __expf(-d2 * s_inv[k]) * s_amp[k];
  }
}

// ---------------------------------------------------------------------------
// Stage 4: per (b,h): G[k][d] = sum_s aff_k[s][k] * v_head[s][d]  (16 x 64)
//                     svec[k] = sum_s aff_k[s][k]
// grid = B*H blocks, 256 threads; thread owns 4 (k,d) cells.
// ---------------------------------------------------------------------------
__global__ void __launch_bounds__(256)
splat_G_kernel(const float* __restrict__ kaff, const float* __restrict__ v,
               float* __restrict__ G, float* __restrict__ svec) {
  const int bh = blockIdx.x;
  const int b = bh / Hq, h = bh % Hq;
  const int tid = threadIdx.x;
  __shared__ float s_aff[16 * Kq];   // 16 s-values x 16 splats
  __shared__ float s_v[16 * DHq];    // 16 s-values x 64 dims

  float acc[4] = {0.f, 0.f, 0.f, 0.f};
  float ssum = 0.f;
  const float* affb = kaff + (size_t)bh * Sq * Kq;
  const float* vb   = v + (size_t)b * Sq * Dq + h * DHq;

  for (int s0 = 0; s0 < Sq; s0 += 16) {
    __syncthreads();
    { int sl = tid >> 4, k = tid & 15;
      s_aff[tid] = affb[(size_t)(s0 + sl) * Kq + k]; }
#pragma unroll
    for (int j = 0; j < 4; ++j) {
      int idx = tid + 256 * j; int sl = idx >> 6, d = idx & 63;
      s_v[idx] = vb[(size_t)(s0 + sl) * Dq + d];
    }
    __syncthreads();
#pragma unroll
    for (int j = 0; j < 4; ++j) {
      int kd = tid + 256 * j; int k = kd >> 6, d = kd & 63;
      float a = 0.f;
#pragma unroll
      for (int sl = 0; sl < 16; ++sl) a += s_aff[sl * Kq + k] * s_v[sl * DHq + d];
      acc[j] += a;
    }
    if (tid < Kq) {
      float a = 0.f;
#pragma unroll
      for (int sl = 0; sl < 16; ++sl) a += s_aff[sl * Kq + tid];
      ssum += a;
    }
  }
#pragma unroll
  for (int j = 0; j < 4; ++j) G[(size_t)bh * 1024 + tid + 256 * j] = acc[j];
  if (tid < Kq) svec[bh * Kq + tid] = ssum;
}

// ---------------------------------------------------------------------------
// Stage 5: out_head[s][d] = (wq[s] . G[:,d]) / (wq[s] . svec + EPS)  -> bf16
// grid = (S/64, H, B), 256 threads; thread covers (d = tid&63) x 16 s values.
// ---------------------------------------------------------------------------
__global__ void __launch_bounds__(256)
splat_out_kernel(const float* __restrict__ wq, const float* __restrict__ G,
                 const float* __restrict__ svec, __bf16* __restrict__ attn_bf) {
  const int h = blockIdx.y, b = blockIdx.z;
  const int bh = b * Hq + h;
  const int s0 = blockIdx.x * 64;
  const int tid = threadIdx.x;
  __shared__ float s_G[Kq * DHq];
  __shared__ float s_s[Kq];
  __shared__ float s_wq[64 * Kq];

#pragma unroll
  for (int j = 0; j < 4; ++j)
    s_G[tid + 256 * j] = G[(size_t)bh * 1024 + tid + 256 * j];
  if (tid < Kq) s_s[tid] = svec[bh * Kq + tid];
#pragma unroll
  for (int j = 0; j < 4; ++j) {
    int idx = tid + 256 * j; int sl = idx >> 4, k = idx & 15;
    s_wq[idx] = wq[((size_t)bh * Sq + s0 + sl) * Kq + k];
  }
  __syncthreads();

  const int d = tid & 63;
  const int slb = tid >> 6;   // 0..3
  for (int i = 0; i < 16; ++i) {
    int sl = slb * 16 + i;
    const float* w = &s_wq[sl * Kq];
    float denom = EPSF, num = 0.f;
#pragma unroll
    for (int k = 0; k < Kq; ++k) {
      denom += w[k] * s_s[k];
      num   += w[k] * s_G[k * DHq + d];
    }
    attn_bf[((size_t)(b * Sq + s0 + sl)) * Dq + h * DHq + d] = (__bf16)(num / denom);
  }
}

// ---------------------------------------------------------------------------
// Host-side launcher
// ---------------------------------------------------------------------------
extern "C" void kernel_launch(void* const* d_in, const int* in_sizes, int n_in,
                              void* d_out, int out_size, void* d_ws, size_t ws_size,
                              hipStream_t stream) {
  const float* x   = (const float*)d_in[0];
  const float* Wq  = (const float*)d_in[1];
  const float* Wk  = (const float*)d_in[2];
  const float* Wv  = (const float*)d_in[3];
  const float* Wo  = (const float*)d_in[4];
  const float* pos = (const float*)d_in[5];
  const float* lsc = (const float*)d_in[6];
  const float* amp = (const float*)d_in[7];
  float* out = (float*)d_out;

  // workspace layout (all offsets 1KB-aligned):
  char* w = (char*)d_ws;
  __bf16* x_bf = (__bf16*)w;                                   // 4096*512 bf16
  __bf16* Wt   = x_bf + (size_t)MROWS * Dq;                    // 4 * 512*512 bf16 (q,k,v,o transposed)
  float*  qkv  = (float*)(Wt + (size_t)4 * Dq * Dq);           // 3 * 4096*512 f32
  float*  q    = qkv;
  float*  k    = qkv + (size_t)MROWS * Dq;
  float*  v    = qkv + (size_t)2 * MROWS * Dq;
  float*  wqa  = qkv + (size_t)3 * MROWS * Dq;                 // B*H*S*K f32
  float*  kaff = wqa + (size_t)Bq * Hq * Sq * Kq;              // B*H*S*K f32
  float*  G    = kaff + (size_t)Bq * Hq * Sq * Kq;             // 16 * 1024 f32
  float*  svec = G + (size_t)Bq * Hq * Kq * DHq;               // 256 f32
  __bf16* attn_bf = (__bf16*)(svec + 256);                     // 4096*512 bf16

  // 1) bf16 conversions
  convert_x_kernel<<<(MROWS * Dq + 255) / 256, 256, 0, stream>>>(
      x, x_bf, MROWS * Dq);
  convert_w_kernel<<<dim3(Dq * Dq / 256, 1, 4), 256, 0, stream>>>(
      Wq, Wk, Wv, Wo, Wt);

  // 2) q,k,v = x @ {Wq,Wk,Wv}  (batched over z)
  wmma_gemm_kernel<<<dim3(Dq / 64, MROWS / 64, 3), 128, 0, stream>>>(
      x_bf, Wt, qkv, (long)Dq * Dq, (long)MROWS * Dq);

  // 3) splat affinities (q -> wq with amp, k -> aff_k)
  splat_affinity_kernel<<<dim3(Sq / 256, Hq, Bq), 256, 0, stream>>>(
      q, pos, lsc, amp, wqa, 1);
  splat_affinity_kernel<<<dim3(Sq / 256, Hq, Bq), 256, 0, stream>>>(
      k, pos, lsc, amp, kaff, 0);

  // 4) rank-16 contraction: G = aff_k^T v, svec = colsum(aff_k)
  splat_G_kernel<<<Bq * Hq, 256, 0, stream>>>(kaff, v, G, svec);

  // 5) normalized output heads -> bf16
  splat_out_kernel<<<dim3(Sq / 64, Hq, Bq), 256, 0, stream>>>(
      wqa, G, svec, attn_bf);

  // 6) final projection: out = attn @ Wo
  wmma_gemm_kernel<<<dim3(Dq / 64, MROWS / 64, 1), 128, 0, stream>>>(
      attn_bf, Wt + (size_t)3 * Dq * Dq, out, 0, 0);
}
